// TFSwinLayer_55336358641806
// MI455X (gfx1250) — matrix operand order repaired and checked by reading, not compiled, not run on hardware
//
#include <hip/hip_runtime.h>
#include <hip/hip_bf16.h>

// ---------------------------------------------------------------------------
// Swin shifted-window transformer layer for MI455X (gfx1250, wave32, WMMA).
// All GEMMs run through v_wmma_f32_16x16x32_bf16 with fp32 accumulation.
// Weights are pre-swizzled into fragment order so every B fragment is two
// global_load_b128; A fragments are two ds_load_b128.
// ---------------------------------------------------------------------------

#define BATCH 64
#define HH 56
#define WW2 56
#define CC 128
#define HEADS 4
#define WSZ 7
#define SHIFT 3
#define HD 32          // head dim
#define NTOK 49        // tokens per window
#define NPAD 64        // padded tokens
#define NW 64          // windows per image
#define MLPD 512

typedef __attribute__((ext_vector_type(16))) __bf16 bf16x16;
typedef __attribute__((ext_vector_type(8)))  float  floatx8;

union FragBF {
    bf16x16 v;
    unsigned short u[16];
    uint4 q[2];
};

static __device__ __forceinline__ unsigned short f2bf(float f) {
    unsigned int u = __float_as_uint(f);
    unsigned int r = u + 0x7FFFu + ((u >> 16) & 1u);  // RNE
    return (unsigned short)(r >> 16);
}

static __device__ __forceinline__ int regid(int x) {
    // shifted-window region id along one axis: [0,49) [49,53) [53,56)
    return x < (HH - WSZ) ? 0 : (x < (HH - SHIFT) ? 1 : 2);
}

// A 16x32 bf16 fragment: elements 0..7 and 8..15 sit 16 elements apart.
// caller passes p = &buf[row*stride + k0 + 8*kh]
static __device__ __forceinline__ void load_frag_gap(FragBF& a, const unsigned short* p) {
    a.q[0] = *(const uint4*)(p);
    a.q[1] = *(const uint4*)(p + 16);
}
// contiguous 16 bf16 (swizzled-B layout)
static __device__ __forceinline__ void load_frag_seq(FragBF& a, const unsigned short* p) {
    a.q[0] = *(const uint4*)(p);
    a.q[1] = *(const uint4*)(p + 8);
}

// ---------------------------------------------------------------------------
// Kernel 0: fp32 -> bf16 weight conversion WITH fragment swizzle.
// dst[((nt*ktn + kt)*32 + lane)*16 + i] = W[(kt*32 + 8*kh + i + (i&8))*Nd + nt*16+col]
// ---------------------------------------------------------------------------
__global__ void cvt_swz_kernel(const float* __restrict__ src,
                               unsigned short* __restrict__ dst, int Kd, int Nd) {
    int e = blockIdx.x * blockDim.x + threadIdx.x;
    if (e >= Kd * Nd) return;
    int i    = e & 15;
    int lane = (e >> 4) & 31;
    int tile = e >> 9;
    int ktn  = Kd >> 5;
    int kt   = tile % ktn;
    int nt   = tile / ktn;
    int kh = lane >> 4, col = lane & 15;
    int sidx = (kt * 32 + 8 * kh + i + (i & 8)) * Nd + nt * 16 + col;
    dst[e] = f2bf(src[sidx]);
}

// B fragment from swizzled weights: 32 bytes contiguous per lane.
static __device__ __forceinline__ void load_b_swz(FragBF& b, const unsigned short* W,
                                                  int tileIdx, int lane) {
    load_frag_seq(b, W + ((size_t)tileIdx << 9) + lane * 16);
}

#define WMMA_BF16(a, b, c) \
    __builtin_amdgcn_wmma_f32_16x16x32_bf16(false, (a).v, false, (b).v, (short)0, (c), false, false)

// ---------------------------------------------------------------------------
// Kernel 1: LN1 + cyclic shift + window partition + QKV projection (WMMA)
// one block per window; 256 threads = 8 waves; LDS: x tile [64][136] bf16
// ---------------------------------------------------------------------------
__global__ __launch_bounds__(256) void qkv_kernel(
    const float* __restrict__ hid, const float* __restrict__ g1,
    const float* __restrict__ b1,
    const unsigned short* __restrict__ wq, const unsigned short* __restrict__ wk,
    const unsigned short* __restrict__ wv,
    const float* __restrict__ qb, const float* __restrict__ kb,
    const float* __restrict__ vb,
    unsigned short* __restrict__ Q, unsigned short* __restrict__ K,
    unsigned short* __restrict__ V)
{
    extern __shared__ char smem[];
    unsigned short* xS = (unsigned short*)smem;  // [64][136]
    const int XS = 136;

    const int w = blockIdx.x;
    const int b = w >> 6, win = w & 63, gh = win >> 3, gw = win & 7;
    const int tid = threadIdx.x, wid = tid >> 5, lane = tid & 31;

    // ---- LN1 + gather (shift + window) into LDS as bf16 ----
    for (int t = wid; t < NPAD; t += 8) {
        if (t < NTOK) {
            int r = t / WSZ, c = t % WSZ;
            int h = (gh * WSZ + r + SHIFT) % HH;
            int x = (gw * WSZ + c + SHIFT) % WW2;
            const float4* src = (const float4*)(hid +
                ((size_t)(b * HH * WW2 + h * WW2 + x)) * CC) + lane;
            float4 v4 = *src;
            float s  = v4.x + v4.y + v4.z + v4.w;
            float s2 = v4.x * v4.x + v4.y * v4.y + v4.z * v4.z + v4.w * v4.w;
            #pragma unroll
            for (int m = 16; m > 0; m >>= 1) {
                s  += __shfl_xor(s,  m, 32);
                s2 += __shfl_xor(s2, m, 32);
            }
            float mean = s * (1.0f / CC);
            float rstd = rsqrtf(s2 * (1.0f / CC) - mean * mean + 1e-5f);
            int ch = lane * 4;
            xS[t * XS + ch + 0] = f2bf((v4.x - mean) * rstd * g1[ch + 0] + b1[ch + 0]);
            xS[t * XS + ch + 1] = f2bf((v4.y - mean) * rstd * g1[ch + 1] + b1[ch + 1]);
            xS[t * XS + ch + 2] = f2bf((v4.z - mean) * rstd * g1[ch + 2] + b1[ch + 2]);
            xS[t * XS + ch + 3] = f2bf((v4.w - mean) * rstd * g1[ch + 3] + b1[ch + 3]);
        } else {
            *(uint2*)&xS[t * XS + lane * 4] = make_uint2(0u, 0u);
        }
    }
    __syncthreads();

    // ---- QKV GEMM: 3 x [64,128]@[128,128]; sel unrolled (const pointers) ----
    const int row = lane & 15, kh = lane >> 4, col = row;
    const float qscale = 0.17677669529663687f;  // 1/sqrt(32)
    #pragma unroll
    for (int sel = 0; sel < 3; ++sel) {
        const unsigned short* W = (sel == 0) ? wq : (sel == 1) ? wk : wv;
        const float* bias       = (sel == 0) ? qb : (sel == 1) ? kb : vb;
        unsigned short* D       = (sel == 0) ? Q  : (sel == 1) ? K  : V;
        const float sc          = (sel == 0) ? qscale : 1.0f;
        for (int it = 0; it < 4; ++it) {
            int id = wid * 4 + it;         // 0..31
            int mt = id >> 3, nt = id & 7;
            floatx8 acc = {};
            #pragma unroll
            for (int kt = 0; kt < 4; ++kt) {
                FragBF a, bfr;
                load_frag_gap(a, xS + (mt * 16 + row) * XS + kt * 32 + 8 * kh);
                load_b_swz(bfr, W, nt * 4 + kt, lane);
                acc = WMMA_BF16(a, bfr, acc);
            }
            int n = nt * 16 + col;
            float bn = bias[n];
            #pragma unroll
            for (int r8 = 0; r8 < 8; ++r8) {
                int rr = mt * 16 + r8 + 8 * kh;
                D[(size_t)w * (NPAD * CC) + rr * CC + n] = f2bf((acc[r8] + bn) * sc);
            }
        }
    }
}

// ---------------------------------------------------------------------------
// Kernel 2: windowed attention, one block per (window, head); 128 thr = 4 waves
// scores = q@k^T (+relbias +shiftmask) -> softmax -> ctx = P@v   (all WMMA)
// V is stored transposed in LDS so the ctx B fragment is two ds_load_b128.
// ---------------------------------------------------------------------------
__global__ __launch_bounds__(128) void attn_kernel(
    const unsigned short* __restrict__ Q, const unsigned short* __restrict__ K,
    const unsigned short* __restrict__ V, const float* __restrict__ relb,
    unsigned short* __restrict__ CTX)
{
    extern __shared__ char smem[];
    const int QS = 40, SS = 72, VTS = 72;
    unsigned short* qS = (unsigned short*)smem;              // 64*40
    unsigned short* kS = qS + NPAD * QS;                     // 64*40
    unsigned short* vT = kS + NPAD * QS;                     // 32*72 (transposed)
    float*          sS = (float*)(vT + HD * VTS);            // 64*72 f32
    unsigned short* pS = (unsigned short*)(sS + NPAD * SS);  // 64*72 bf16

    const int bid = blockIdx.x;
    const int w = bid >> 2, hd = bid & 3;
    const int win = w & 63, gh = win >> 3, gw = win & 7;
    const int tid = threadIdx.x, wid = tid >> 5, lane = tid & 31;

    // ---- stage q,k (row-major) and v (transposed) into LDS ----
    {
        int t = tid >> 1, half = tid & 1;
        size_t base = (size_t)w * (NPAD * CC) + (size_t)t * CC + hd * HD + half * 16;
        int doff = t * QS + half * 16;
        uint4 q0 = *(const uint4*)(Q + base), q1 = *(const uint4*)(Q + base + 8);
        uint4 k0 = *(const uint4*)(K + base), k1 = *(const uint4*)(K + base + 8);
        *(uint4*)(qS + doff)     = q0;
        *(uint4*)(qS + doff + 8) = q1;
        *(uint4*)(kS + doff)     = k0;
        *(uint4*)(kS + doff + 8) = k1;
        const unsigned short* vsrc = V + base;
        #pragma unroll
        for (int j = 0; j < 16; ++j) vT[(half * 16 + j) * VTS + t] = vsrc[j];
    }
    __syncthreads();

    const int row = lane & 15, kh = lane >> 4, col = row;

    // ---- scores: 16 tiles of 16x16, K=32 (one WMMA each) ----
    for (int it = 0; it < 4; ++it) {
        int tt = wid * 4 + it;
        int mt = tt >> 2, nt = tt & 3;
        FragBF a, bb;
        load_frag_gap(a, qS + (mt * 16 + row) * QS + 8 * kh);
        load_frag_gap(bb, kS + (nt * 16 + col) * QS + 8 * kh);  // k^T
        floatx8 acc = {};
        acc = WMMA_BF16(a, bb, acc);

        int n = nt * 16 + col;
        #pragma unroll
        for (int r8 = 0; r8 < 8; ++r8) {
            int mr = mt * 16 + r8 + 8 * kh;
            float s;
            if (n >= NTOK) {
                s = -1e9f;  // pad column
            } else {
                s = acc[r8];
                if (mr < NTOK) {
                    int ri = mr / WSZ, ci = mr % WSZ;
                    int rj = n / WSZ,  cj = n % WSZ;
                    int idx = (ri - rj + WSZ - 1) * (2 * WSZ - 1) + (ci - cj + WSZ - 1);
                    s += relb[idx * HEADS + hd];
                    int idI = 3 * regid(gh * WSZ + ri) + regid(gw * WSZ + ci);
                    int idJ = 3 * regid(gh * WSZ + rj) + regid(gw * WSZ + cj);
                    if (idI != idJ) s -= 100.0f;
                }
            }
            sS[mr * SS + n] = s;
        }
    }
    __syncthreads();

    // ---- softmax over 64 cols (pad cols hold -1e9) ----
    for (int rr = 0; rr < 16; ++rr) {
        int mr = wid * 16 + rr;
        float s0 = sS[mr * SS + lane];
        float s1 = sS[mr * SS + 32 + lane];
        float mx = fmaxf(s0, s1);
        #pragma unroll
        for (int m = 16; m > 0; m >>= 1) mx = fmaxf(mx, __shfl_xor(mx, m, 32));
        float e0 = __expf(s0 - mx), e1 = __expf(s1 - mx);
        float sm = e0 + e1;
        #pragma unroll
        for (int m = 16; m > 0; m >>= 1) sm += __shfl_xor(sm, m, 32);
        float inv = 1.0f / sm;
        pS[mr * SS + lane]      = f2bf(e0 * inv);
        pS[mr * SS + 32 + lane] = f2bf(e1 * inv);
    }
    __syncthreads();

    // ---- ctx = P[64x64] @ v[64x32]: 8 tiles, K=64 (2 WMMA steps) ----
    for (int it = 0; it < 2; ++it) {
        int tt = wid * 2 + it;
        int mt = tt >> 1, nt = tt & 1;
        floatx8 acc = {};
        #pragma unroll
        for (int kt = 0; kt < 2; ++kt) {
            FragBF a, bb;
            load_frag_gap(a, pS + (mt * 16 + row) * SS + kt * 32 + 8 * kh);
            load_frag_gap(bb, vT + (nt * 16 + col) * VTS + kt * 32 + 8 * kh);
            acc = WMMA_BF16(a, bb, acc);
        }
        int n = nt * 16 + col;
        #pragma unroll
        for (int r8 = 0; r8 < 8; ++r8) {
            int mr = mt * 16 + r8 + 8 * kh;
            CTX[(size_t)w * (NPAD * CC) + mr * CC + hd * HD + n] = f2bf(acc[r8]);
        }
    }
}

// ---------------------------------------------------------------------------
// Kernel 3: output projection + window reverse + un-shift + residual -> d_out
// ---------------------------------------------------------------------------
__global__ __launch_bounds__(256) void proj_kernel(
    const unsigned short* __restrict__ CTX, const unsigned short* __restrict__ wp,
    const float* __restrict__ pb, const float* __restrict__ hid,
    float* __restrict__ out)
{
    extern __shared__ char smem[];
    unsigned short* cS = (unsigned short*)smem;  // [64][136]
    const int XS = 136;

    const int w = blockIdx.x;
    const int b = w >> 6, win = w & 63, gh = win >> 3, gw = win & 7;
    const int tid = threadIdx.x, wid = tid >> 5, lane = tid & 31;

    {
        int t = tid >> 2, qq = tid & 3;
        const unsigned short* src = CTX + (size_t)w * (NPAD * CC) + (size_t)t * CC + qq * 32;
        unsigned short* dst = cS + t * XS + qq * 32;
        #pragma unroll
        for (int j = 0; j < 4; ++j)
            *(uint4*)(dst + j * 8) = *(const uint4*)(src + j * 8);
    }
    __syncthreads();

    const int row = lane & 15, kh = lane >> 4, col = row;
    for (int it = 0; it < 4; ++it) {
        int id = wid * 4 + it;
        int mt = id >> 3, nt = id & 7;
        floatx8 acc = {};
        #pragma unroll
        for (int kt = 0; kt < 4; ++kt) {
            FragBF a, bb;
            load_frag_gap(a, cS + (mt * 16 + row) * XS + kt * 32 + 8 * kh);
            load_b_swz(bb, wp, nt * 4 + kt, lane);
            acc = WMMA_BF16(a, bb, acc);
        }
        int n = nt * 16 + col;
        float bn = pb[n];
        #pragma unroll
        for (int r8 = 0; r8 < 8; ++r8) {
            int mr = mt * 16 + r8 + 8 * kh;
            if (mr < NTOK) {
                int r7 = mr / WSZ, c7 = mr % WSZ;
                int h = (gh * WSZ + r7 + SHIFT) % HH;
                int x = (gw * WSZ + c7 + SHIFT) % WW2;
                size_t di = ((size_t)(b * HH * WW2 + h * WW2 + x)) * CC + n;
                out[di] = hid[di] + acc[r8] + bn;
            }
        }
    }
}

// ---------------------------------------------------------------------------
// Kernel 4: fused LN2 + FC1 + GELU + FC2 + residual.  64 rows per block.
// GELU intermediate [64x512] lives entirely in LDS (bf16) — never hits HBM.
// ---------------------------------------------------------------------------
__global__ __launch_bounds__(256) void mlp_kernel(
    float* __restrict__ xio, const float* __restrict__ g2,
    const float* __restrict__ b2,
    const unsigned short* __restrict__ w1, const float* __restrict__ b1v,
    const unsigned short* __restrict__ w2, const float* __restrict__ b2v)
{
    extern __shared__ char smem[];
    const int XS = 136, HS = 528;
    unsigned short* lnS = (unsigned short*)smem;   // [64][136]
    unsigned short* hS  = lnS + 64 * XS;           // [64][528]

    const size_t rows0 = (size_t)blockIdx.x * 64;
    const int tid = threadIdx.x, wid = tid >> 5, lane = tid & 31;

    // ---- LN2 ----
    for (int t = wid; t < 64; t += 8) {
        const float4* src = (const float4*)(xio + (rows0 + t) * CC) + lane;
        float4 v4 = *src;
        float s  = v4.x + v4.y + v4.z + v4.w;
        float s2 = v4.x * v4.x + v4.y * v4.y + v4.z * v4.z + v4.w * v4.w;
        #pragma unroll
        for (int m = 16; m > 0; m >>= 1) {
            s  += __shfl_xor(s,  m, 32);
            s2 += __shfl_xor(s2, m, 32);
        }
        float mean = s * (1.0f / CC);
        float rstd = rsqrtf(s2 * (1.0f / CC) - mean * mean + 1e-5f);
        int ch = lane * 4;
        lnS[t * XS + ch + 0] = f2bf((v4.x - mean) * rstd * g2[ch + 0] + b2[ch + 0]);
        lnS[t * XS + ch + 1] = f2bf((v4.y - mean) * rstd * g2[ch + 1] + b2[ch + 1]);
        lnS[t * XS + ch + 2] = f2bf((v4.z - mean) * rstd * g2[ch + 2] + b2[ch + 2]);
        lnS[t * XS + ch + 3] = f2bf((v4.w - mean) * rstd * g2[ch + 3] + b2[ch + 3]);
    }
    __syncthreads();

    const int row = lane & 15, kh = lane >> 4, col = row;

    // ---- FC1 + exact GELU: [64,128]@[128,512]; 128 tiles / 8 waves ----
    for (int it = 0; it < 16; ++it) {
        int id = wid * 16 + it;
        int mt = id >> 5, nt = id & 31;
        floatx8 acc = {};
        #pragma unroll
        for (int kt = 0; kt < 4; ++kt) {
            FragBF a, bb;
            load_frag_gap(a, lnS + (mt * 16 + row) * XS + kt * 32 + 8 * kh);
            load_b_swz(bb, w1, nt * 4 + kt, lane);
            acc = WMMA_BF16(a, bb, acc);
        }
        int n = nt * 16 + col;
        float bn = b1v[n];
        #pragma unroll
        for (int r8 = 0; r8 < 8; ++r8) {
            int mr = mt * 16 + r8 + 8 * kh;
            float v = acc[r8] + bn;
            float gv = 0.5f * v * (1.0f + erff(v * 0.7071067811865475f));
            hS[mr * HS + n] = f2bf(gv);
        }
    }
    __syncthreads();

    // ---- FC2 + residual: [64,512]@[512,128]; 32 tiles / 8 waves ----
    for (int it = 0; it < 4; ++it) {
        int id = wid * 4 + it;
        int mt = id >> 3, nt = id & 7;
        floatx8 acc = {};
        #pragma unroll
        for (int kt = 0; kt < 16; ++kt) {
            FragBF a, bb;
            load_frag_gap(a, hS + (mt * 16 + row) * HS + kt * 32 + 8 * kh);
            load_b_swz(bb, w2, nt * 16 + kt, lane);
            acc = WMMA_BF16(a, bb, acc);
        }
        int n = nt * 16 + col;
        float bn = b2v[n];
        #pragma unroll
        for (int r8 = 0; r8 < 8; ++r8) {
            int mr = mt * 16 + r8 + 8 * kh;
            size_t di = (rows0 + mr) * CC + n;
            xio[di] = xio[di] + acc[r8] + bn;
        }
    }
}

// ---------------------------------------------------------------------------
// launcher
// ---------------------------------------------------------------------------
extern "C" void kernel_launch(void* const* d_in, const int* in_sizes, int n_in,
                              void* d_out, int out_size, void* d_ws, size_t ws_size,
                              hipStream_t stream) {
    (void)in_sizes; (void)n_in; (void)out_size; (void)ws_size;

    const float* hid   = (const float*)d_in[0];
    const float* ln1g  = (const float*)d_in[1];
    const float* ln1b  = (const float*)d_in[2];
    const float* q_w   = (const float*)d_in[3];
    const float* q_b   = (const float*)d_in[4];
    const float* k_w   = (const float*)d_in[5];
    const float* k_b   = (const float*)d_in[6];
    const float* v_w   = (const float*)d_in[7];
    const float* v_b   = (const float*)d_in[8];
    const float* p_w   = (const float*)d_in[9];
    const float* p_b   = (const float*)d_in[10];
    const float* relb  = (const float*)d_in[11];
    const float* ln2g  = (const float*)d_in[12];
    const float* ln2b  = (const float*)d_in[13];
    const float* fc1_w = (const float*)d_in[14];
    const float* fc1_b = (const float*)d_in[15];
    const float* fc2_w = (const float*)d_in[16];
    const float* fc2_b = (const float*)d_in[17];
    float* out = (float*)d_out;

    // workspace layout (bytes)
    char* ws = (char*)d_ws;
    const size_t SQ = (size_t)CC * CC * 2;            // 32 KB per 128x128 bf16
    unsigned short* wq   = (unsigned short*)(ws);
    unsigned short* wk   = (unsigned short*)(ws + SQ);
    unsigned short* wv   = (unsigned short*)(ws + 2 * SQ);
    unsigned short* wp   = (unsigned short*)(ws + 3 * SQ);
    unsigned short* wfc1 = (unsigned short*)(ws + 4 * SQ);                 // 128x512
    unsigned short* wfc2 = (unsigned short*)(ws + 4 * SQ + (size_t)CC * MLPD * 2);
    size_t act0 = 4 * SQ + 2 * (size_t)CC * MLPD * 2;                      // 393216
    const size_t ABUF = (size_t)BATCH * NW * NPAD * CC * 2;                // 64 MB
    unsigned short* Qb = (unsigned short*)(ws + act0);
    unsigned short* Kb = (unsigned short*)(ws + act0 + ABUF);
    unsigned short* Vb = (unsigned short*)(ws + act0 + 2 * ABUF);
    unsigned short* Cb = (unsigned short*)(ws + act0 + 3 * ABUF);

    // 0) weights -> bf16, swizzled into WMMA B-fragment order
    auto cvt = [&](const float* s, unsigned short* d, int Kd, int Nd) {
        int n = Kd * Nd;
        cvt_swz_kernel<<<(n + 255) / 256, 256, 0, stream>>>(s, d, Kd, Nd);
    };
    cvt(q_w, wq, CC, CC);
    cvt(k_w, wk, CC, CC);
    cvt(v_w, wv, CC, CC);
    cvt(p_w, wp, CC, CC);
    cvt(fc1_w, wfc1, CC, MLPD);
    cvt(fc2_w, wfc2, MLPD, CC);

    const int NWIN = BATCH * NW;  // 4096

    // 1) LN1 + shift + window + QKV
    qkv_kernel<<<NWIN, 256, 64 * 136 * 2, stream>>>(
        hid, ln1g, ln1b, wq, wk, wv, q_b, k_b, v_b, Qb, Kb, Vb);

    // 2) attention per (window, head)
    size_t attn_lds = 2 * (size_t)NPAD * 40 * 2 + (size_t)HD * 72 * 2 +
                      (size_t)NPAD * 72 * 4 + (size_t)NPAD * 72 * 2;  // 42496 B
    attn_kernel<<<NWIN * HEADS, 128, attn_lds, stream>>>(Qb, Kb, Vb, relb, Cb);

    // 3) proj + reverse + unshift + residual -> d_out (= x)
    proj_kernel<<<NWIN, 256, 64 * 136 * 2, stream>>>(Cb, wp, p_b, hid, out);

    // 4) LN2 + FC1 + GELU + FC2 + residual (in-place on d_out)
    size_t mlp_lds = 64 * 136 * 2 + 64 * 528 * 2;  // 84992 B (fits 320KB WGP LDS)
    mlp_kernel<<<(BATCH * HH * WW2) / 64, 256, mlp_lds, stream>>>(
        out, ln2g, ln2b, wfc1, fc1_b, wfc2, fc2_b);
}
